// ChooseDest_37555194036651
// MI455X (gfx1250) — matrix operand order repaired and checked by reading, not compiled, not run on hardware
//
#include <hip/hip_runtime.h>

typedef _Float16 h8   __attribute__((ext_vector_type(8)));
typedef _Float16 v16h __attribute__((ext_vector_type(16)));
typedef float    v8f  __attribute__((ext_vector_type(8)));

#define BB    256
#define NMAXN 512
#define HHH   256
#define CCC   64
#define INDIM 576   // 2H + C
#define HIDD  512
#define MT    64    // rows per tile
#define TILES 8     // 512 / 64
#define XS    584   // LDS stride (halves) for X buffer, padded
#define HS    520   // LDS stride (halves) for hidden buffers, padded
#define NEGV  (-1000000000.0f)

// ---------------- prep: transpose + f32->f16 weights ----------------
__global__ void prep_weights_kernel(const float* __restrict__ W1,
                                    const float* __restrict__ W2,
                                    _Float16* __restrict__ W1T,
                                    _Float16* __restrict__ W2T) {
    int idx = blockIdx.x * blockDim.x + threadIdx.x;
    if (idx < HIDD * INDIM) {
        int h = idx / INDIM, i = idx - h * INDIM;
        W1T[idx] = (_Float16)W1[i * HIDD + h];          // W1T[h][i], k-contiguous
    } else {
        int j = idx - HIDD * INDIM;
        if (j < HIDD * HIDD) {
            int o = j / HIDD, i = j - o * HIDD;
            W2T[j] = (_Float16)W2[i * HIDD + o];        // W2T[o][i]
        }
    }
}

// Build a v16h A/B fragment from two non-contiguous 16B runs.
__device__ __forceinline__ v16h frag_from2(const _Float16* p0, const _Float16* p1) {
    h8 lo = *(const h8*)p0;
    h8 hi = *(const h8*)p1;
    return __builtin_shufflevector(lo, hi, 0,1,2,3,4,5,6,7,8,9,10,11,12,13,14,15);
}

// One dense layer: Out = relu(A(64xK, LDS f16) * W(KxN col-major global f16) + bias)
// Wave 'wid' owns columns [wid*64, wid*64+64); computes all 64 rows.
template<int K>
__device__ __forceinline__ void mlp_layer(const _Float16* __restrict__ As, int sA,
                                          const _Float16* __restrict__ Wt,
                                          const float* __restrict__ bias,
                                          _Float16* __restrict__ Os, int sO,
                                          int wid, int lane) {
    const int half = lane >> 4;        // 0: K 0..15, 1: K 16..31
    const int mrow = lane & 15;
    const int colwave = wid * 64;

    v8f acc[4][4] = {};                // [rowblock][nblock], 16x16 f32 tiles

    const _Float16* a0 = As + mrow * sA + half * 16;             // row = rb*16 + mrow
    const _Float16* wp0 = Wt + (size_t)(colwave + mrow) * K + half * 16;

    for (int ks = 0; ks < K / 32; ++ks) {
        // A fragments for the four 16-row blocks (16 contiguous K per lane, matching B)
        v16h a[4];
#pragma unroll
        for (int rb = 0; rb < 4; ++rb) {
            const _Float16* p = a0 + rb * 16 * sA + ks * 32;
            a[rb] = frag_from2(p, p + 8);
        }
#pragma unroll
        for (int nb = 0; nb < 4; ++nb) {
            // B fragment: col = lane&15 (+nb*16), 16 contiguous K per lane (same k order as A)
            v16h b = *(const v16h*)(wp0 + (size_t)nb * 16 * K + ks * 32);
#pragma unroll
            for (int rb = 0; rb < 4; ++rb) {
                acc[rb][nb] = __builtin_amdgcn_wmma_f32_16x16x32_f16(
                    false, a[rb], false, b, (short)0, acc[rb][nb], false, false);
            }
        }
    }

    // epilogue: bias + relu, f32 -> f16, store to LDS row-major
#pragma unroll
    for (int nb = 0; nb < 4; ++nb) {
        int col = colwave + nb * 16 + mrow;
        float bv = bias[col];
#pragma unroll
        for (int rb = 0; rb < 4; ++rb) {
#pragma unroll
            for (int v = 0; v < 8; ++v) {
                float x = acc[rb][nb][v] + bv;          // C layout: M = v + 8*half
                x = x > 0.0f ? x : 0.0f;
                Os[(rb * 16 + v + 8 * half) * sO + col] = (_Float16)x;
            }
        }
    }
}

// ---------------- fused MLP over a 64-row tile ----------------
__global__ __launch_bounds__(256)
void mlp_tile_kernel(const float* __restrict__ nodes_hv,
                     const float* __restrict__ class_cond,
                     const _Float16* __restrict__ W1T, const float* __restrict__ b1,
                     const _Float16* __restrict__ W2T, const float* __restrict__ b2,
                     const float* __restrict__ W3,
                     const int* __restrict__ dirns,
                     const int* __restrict__ node_counts,
                     float* __restrict__ scores_ws) {
    extern __shared__ char smem[];
    _Float16* bufX = (_Float16*)smem;                          // 64 x XS halves
    _Float16* bufH = (_Float16*)(smem + MT * XS * 2);          // 64 x HS halves
    float* partial = (float*)(smem + MT * XS * 2 + MT * HS * 2); // 64 x 4

    const int b    = blockIdx.x / TILES;
    const int tile = blockIdx.x % TILES;
    const int tid  = threadIdx.x;

    const int nc   = node_counts[b];
    const int dirn = dirns[b];
    const int srow = nc - 1;

    const float* nodeb = nodes_hv + (size_t)b * NMAXN * HHH;
    const float* ccb   = class_cond + (size_t)b * CCC;

    // ---- build X tile (select + concat + f32->f16) ----
    for (int idx = tid; idx < MT * INDIM; idx += 256) {
        int r = idx / INDIM;
        int c = idx - r * INDIM;
        int d = tile * MT + r;           // d==511 is a pad row, masked later
        float val;
        if (c < 2 * HHH) {
            int cf = (c < HHH) ? c : (c - HHH);
            float cand = nodeb[(size_t)d * HHH + cf];
            float src  = nodeb[(size_t)srow * HHH + cf];
            bool first = (c < HHH);
            // side_a = dirn ? cand : src ; side_b = dirn ? src : cand
            val = (first == (dirn != 0)) ? cand : src;
        } else {
            val = ccb[c - 2 * HHH];
        }
        bufX[r * XS + c] = (_Float16)val;
    }
    __syncthreads();

    const int wid  = tid >> 5;
    const int lane = tid & 31;

    mlp_layer<INDIM>(bufX, XS, W1T, b1, bufH, HS, wid, lane);   // h1 -> bufH
    __syncthreads();
    mlp_layer<HIDD>(bufH, HS, W2T, b2, bufX, HS, wid, lane);    // h2 -> bufX
    __syncthreads();

    // ---- layer 3: scores = h2 . W3   (b3 cancels in log_softmax) ----
    {
        int row  = tid >> 2;
        int part = tid & 3;
        const _Float16* h2 = bufX + row * HS + part * 128;
        const float* w3 = W3 + part * 128;
        float s = 0.0f;
#pragma unroll
        for (int k8 = 0; k8 < 16; ++k8) {
            h8 hv = *(const h8*)(h2 + k8 * 8);
#pragma unroll
            for (int j = 0; j < 8; ++j) s += (float)hv[j] * w3[k8 * 8 + j];
        }
        partial[row * 4 + part] = s;
    }
    __syncthreads();
    if (tid < MT) {
        float s = partial[tid * 4] + partial[tid * 4 + 1]
                + partial[tid * 4 + 2] + partial[tid * 4 + 3];
        int d = tile * MT + tid;
        bool valid = d < (nc - 1);
        scores_ws[(size_t)b * NMAXN + d] = valid ? s : NEGV;
    }
}

// ---------------- per-batch log-softmax + NLL ----------------
__global__ void softmax_loss_kernel(const float* __restrict__ scores_ws,
                                    const int* __restrict__ dests,
                                    float* __restrict__ out) {
    __shared__ float red[256];
    int b = blockIdx.x;
    int t = threadIdx.x;
    const float* s = scores_ws + (size_t)b * NMAXN;
    float a0 = s[t], a1 = s[t + 256];

    red[t] = fmaxf(a0, a1);
    __syncthreads();
    for (int off = 128; off > 0; off >>= 1) {
        if (t < off) red[t] = fmaxf(red[t], red[t + off]);
        __syncthreads();
    }
    float mx = red[0];
    __syncthreads();

    red[t] = expf(a0 - mx) + expf(a1 - mx);
    __syncthreads();
    for (int off = 128; off > 0; off >>= 1) {
        if (t < off) red[t] += red[t + off];
        __syncthreads();
    }
    if (t == 0) {
        float lse = mx + logf(red[0]);
        out[b] = -(s[dests[b]] - lse);
    }
}

extern "C" void kernel_launch(void* const* d_in, const int* in_sizes, int n_in,
                              void* d_out, int out_size, void* d_ws, size_t ws_size,
                              hipStream_t stream) {
    const float* nodes_hv   = (const float*)d_in[0];
    const float* class_cond = (const float*)d_in[1];
    const float* W1 = (const float*)d_in[2];
    const float* b1 = (const float*)d_in[3];
    const float* W2 = (const float*)d_in[4];
    const float* b2 = (const float*)d_in[5];
    const float* W3 = (const float*)d_in[6];
    // d_in[7] = b3: constant offset cancels exactly in log_softmax
    const int* dirns       = (const int*)d_in[8];
    const int* node_counts = (const int*)d_in[9];
    const int* dests       = (const int*)d_in[10];

    char* ws = (char*)d_ws;
    _Float16* W1T   = (_Float16*)(ws);             // 512*576*2 = 589824 B
    _Float16* W2T   = (_Float16*)(ws + 589824);    // 512*512*2 = 524288 B
    float*    score = (float*)(ws + 1114112);      // 256*512*4 = 524288 B

    prep_weights_kernel<<<(HIDD * INDIM + HIDD * HIDD) / 256, 256, 0, stream>>>(
        W1, W2, W1T, W2T);

    size_t smem = (size_t)MT * XS * 2 + (size_t)MT * HS * 2 + MT * 4 * sizeof(float);
    mlp_tile_kernel<<<BB * TILES, 256, smem, stream>>>(
        nodes_hv, class_cond, W1T, b1, W2T, b2, W3, dirns, node_counts, score);

    softmax_loss_kernel<<<BB, 256, 0, stream>>>(score, dests, (float*)d_out);
}
